// FastAttention_12532714569857
// MI455X (gfx1250) — matrix operand order repaired and checked by reading, not compiled, not run on hardware
//
#include <hip/hip_runtime.h>

// Performer (FAVOR+) linear attention for MI455X (gfx1250, wave32, WMMA).
// B=4, S=8192, H=8, D=64, M=64.  All GEMMs mapped to v_wmma_f32_16x16x32_f16.
// proj and per-bh state staged into LDS with the Tensor Data Mover (TDM).
// The 64^(-1/4) normalizer is folded into the proj fragments; h = 0.5*c^2*||x||^2.

typedef __attribute__((ext_vector_type(16))) _Float16 v16h;
typedef __attribute__((ext_vector_type(8)))  _Float16 v8h;
typedef __attribute__((ext_vector_type(8)))  float    v8f;

#define SS 8192
#define HHN 8
#define DD 64
#define NBH 32
#define ROWSTRIDE (HHN * DD)          // floats between consecutive seq rows (fixed b,h)
#define BUFSLICE 4160                 // per-bh state: 64x64 numerator + 64 denominator
#define SCALE 0.35355339059327373f    // 64^(-1/4)
#define HSCL 0.0625f                  // 0.5 * SCALE^2
#define RSQM 0.125f                   // 1/sqrt(64)
#define EPSV 1e-6f

#if defined(__gfx1250__) && __has_builtin(__builtin_amdgcn_tensor_load_to_lds)
#define USE_TDM 1
#else
#define USE_TDM 0
#endif

#if USE_TDM
typedef __attribute__((ext_vector_type(4))) unsigned int v4u;
typedef __attribute__((ext_vector_type(8))) int          v8i;
typedef __attribute__((ext_vector_type(4))) int          v4i;

// 1-D contiguous copy of nelem f32 from global to LDS via the Tensor Data Mover.
// D# packing per CDNA5 ISA.
__device__ __forceinline__ void tdm_copy_1d(const float* gsrc, void* lds_dst,
                                            unsigned int nelem) {
  unsigned long long ga = (unsigned long long)(uintptr_t)gsrc;
  unsigned int la = (unsigned int)(uintptr_t)lds_dst;  // flat low 32 bits == LDS byte addr
  v4u g0;
  g0.x = 1u;                                              // count=1, user mode
  g0.y = la;                                              // lds_addr
  g0.z = (unsigned int)ga;                                // global_addr[31:0]
  g0.w = (unsigned int)((ga >> 32) & 0x1FFFFFFu) | (2u << 30);  // addr[56:32] | type=2
  v8i g1;
  g1[0] = (int)(2u << 16);                                // data_size=4B, mask=0
  g1[1] = (int)((nelem & 0xFFFFu) << 16);                 // tensor_dim0 lo16
  g1[2] = (int)((nelem >> 16) | (1u << 16));              // tensor_dim0 hi16 | tensor_dim1=1
  g1[3] = (int)((nelem & 0xFFFFu) << 16);                 // tile_dim0 = nelem
  g1[4] = 1;                                              // tile_dim1=1, tile_dim2=0
  g1[5] = (int)nelem;                                     // tensor_dim0_stride lo32
  g1[6] = 0;
  g1[7] = 0;
  v4i gz = {0, 0, 0, 0};
#if __has_include(<hip/amd_detail/amd_gfx1250_TDM.h>)
  v8i z8 = {0, 0, 0, 0, 0, 0, 0, 0};
  __builtin_amdgcn_tensor_load_to_lds(g0, g1, gz, gz, z8, 0);   // clang-23 lane (6 args)
#else
  __builtin_amdgcn_tensor_load_to_lds(g0, g1, gz, gz, 0);       // ROCm 7.2 lane (5 args)
#endif
}
#endif  // USE_TDM

__device__ __forceinline__ v8f wmma16(v16h a, v16h b, v8f c) {
  return __builtin_amdgcn_wmma_f32_16x16x32_f16(false, a, false, b, (short)0, c,
                                                false, false);
}

// broadcast value from lane `src` (0..31) using ds_bpermute (no LDS memory)
__device__ __forceinline__ float lane_bcast(float x, int src) {
  return __int_as_float(__builtin_amdgcn_ds_bpermute(src << 2, __float_as_int(x)));
}

// B-fragment of SCALE * proj^T (32x16): lane l holds col mf+l%16, K=d=32*s+8*g+i (+16 for halves 8..15).
__device__ __forceinline__ v16h load_projT_frag(const float* proj, int s, int mf,
                                                int l) {
  const int gg = l >> 4;
  v16h bfr;
#pragma unroll
  for (int hh = 0; hh < 16; ++hh)
    bfr[hh] = (_Float16)(proj[(mf + (l & 15)) * DD + 32 * s + (hh < 8 ? 8 * gg + hh : 8 + 8 * gg + hh)] * SCALE);
  return bfr;
}

// A-fragment (16x32 f16) of a RAW input row; accumulates raw per-lane sum of squares.
// lane l (g=l>>4): halves 0..7 -> d = 32*s + 8*g + i ; halves 8..15 -> +16.
__device__ __forceinline__ v16h load_x_frag(const float* __restrict__ xrow,
                                            int s, int g, float& ssq) {
  const float4* p0 = (const float4*)(xrow + 32 * s + 8 * g);
  const float4* p1 = (const float4*)(xrow + 32 * s + 8 * g + 16);
  float4 a0 = p0[0], a1 = p0[1], b0 = p1[0], b1 = p1[1];
  float lo[8] = {a0.x, a0.y, a0.z, a0.w, a1.x, a1.y, a1.z, a1.w};
  float hi[8] = {b0.x, b0.y, b0.z, b0.w, b1.x, b1.y, b1.z, b1.w};
  v16h a;
#pragma unroll
  for (int i = 0; i < 8; ++i) {
    ssq += lo[i] * lo[i] + hi[i] * hi[i];
    a[i] = (_Float16)lo[i];
    a[8 + i] = (_Float16)hi[i];
  }
  return a;
}

__global__ void zerok(float* __restrict__ p, int n) {
  int i = blockIdx.x * blockDim.x + threadIdx.x;
  if (i < n) p[i] = 0.0f;
}

// ---------------- Pass 1: state[bh] = k'^T @ [v | 1]  (f32 atomics) ------------
// wave task: (bh, 128-row seq block, feature half). 4 iters x 32 seq rows.
// All 8 waves of a block share one bh; proj staged once per block via TDM.
__global__ void __launch_bounds__(256)
pass1(const float* __restrict__ ks, const float* __restrict__ vs,
      const float* __restrict__ proj, float* __restrict__ bufc) {
#if USE_TDM
  __shared__ __align__(16) float sproj[4096];
#endif
  const int l = threadIdx.x & 31;
  const int wv = threadIdx.x >> 5;
  const int w = blockIdx.x * 8 + wv;
  const int fh = w & 1;            // feature half: rows 0..31 or 32..63 of state
  const int sb = (w >> 1) & 63;    // 64 seq blocks of 128 rows
  const int bh = w >> 7;           // 0..31 (uniform within the block)
  const int b = bh >> 3, h = bh & 7;
  const int g = l >> 4, r = l & 15;
  const int fbase = fh * 32;

#if USE_TDM
  if (wv == 0) {
    tdm_copy_1d(proj, sproj, 4096);
    __builtin_amdgcn_s_wait_tensorcnt(0);
  }
  __syncthreads();
  const float* psrc = sproj;
#else
  const float* psrc = proj;
#endif

  v16h pj[2][2];
#pragma unroll
  for (int s = 0; s < 2; ++s)
#pragma unroll
    for (int f = 0; f < 2; ++f)
      pj[s][f] = load_projT_frag(psrc, s, fbase + 16 * f, l);

  v8f macc[2][4];
#pragma unroll
  for (int f = 0; f < 2; ++f)
#pragma unroll
    for (int n = 0; n < 4; ++n) macc[f][n] = (v8f){};
  float fsum[2] = {0.0f, 0.0f};

  const float* kbase = ks + (((size_t)b * SS) * HHN + h) * DD;
  const float* vbase = vs + (((size_t)b * SS) * HHN + h) * DD;

  for (int it = 0; it < 4; ++it) {
    const int s0 = sb * 128 + it * 32;
    if (it < 3) {  // gfx1250 global_prefetch for next iteration
      __builtin_prefetch(kbase + (size_t)(s0 + 32 + r) * ROWSTRIDE, 0, 1);
      __builtin_prefetch(vbase + (size_t)(s0 + 32 + l) * ROWSTRIDE, 0, 1);
    }
    float e[2][2][8];
#pragma unroll
    for (int t = 0; t < 2; ++t) {
      const float* xrow = kbase + (size_t)(s0 + 16 * t + r) * ROWSTRIDE;
      float ssq = 0.0f;
      v16h a0 = load_x_frag(xrow, 0, g, ssq);
      v16h a1 = load_x_frag(xrow, 1, g, ssq);
      const float hfull = HSCL * (ssq + lane_bcast(ssq, l ^ 16));
      v8f acc[2];
#pragma unroll
      for (int f = 0; f < 2; ++f) {
        v8f c = {};
        c = wmma16(a0, pj[0][f], c);
        c = wmma16(a1, pj[1][f], c);
        acc[f] = c;
      }
      float hrow[8];
#pragma unroll
      for (int j = 0; j < 8; ++j) hrow[j] = lane_bcast(hfull, j + 8 * g);
#pragma unroll
      for (int f = 0; f < 2; ++f)
#pragma unroll
        for (int j = 0; j < 8; ++j) {
          float v = __expf(acc[f][j] - hrow[j]) * RSQM;
          e[t][f][j] = v;
          fsum[f] += v;
        }
    }
    // k'^T A-fragments are free: D(seq x feat) layout == A(feat x seq) layout.
    v16h ak[2];
#pragma unroll
    for (int f = 0; f < 2; ++f)
#pragma unroll
      for (int j = 0; j < 8; ++j) {
        ak[f][j] = (_Float16)e[0][f][j];
        ak[f][8 + j] = (_Float16)e[1][f][j];
      }
#pragma unroll
    for (int n = 0; n < 4; ++n) {
      const float* vrow = vbase + (size_t)s0 * ROWSTRIDE + 16 * n + r;
      v16h bv;
#pragma unroll
      for (int i = 0; i < 16; ++i) bv[i] = (_Float16)vrow[(size_t)(i < 8 ? 8 * g + i : 8 + 8 * g + i) * ROWSTRIDE];
#pragma unroll
      for (int f = 0; f < 2; ++f) macc[f][n] = wmma16(ak[f], bv, macc[f][n]);
    }
  }

  float* slice = bufc + (size_t)bh * BUFSLICE;
#pragma unroll
  for (int f = 0; f < 2; ++f) {
    const int m0 = fbase + 16 * f;
#pragma unroll
    for (int n = 0; n < 4; ++n)
#pragma unroll
      for (int j = 0; j < 8; ++j)
        atomicAdd(&slice[(m0 + j + 8 * g) * 64 + 16 * n + r], macc[f][n][j]);
    atomicAdd(&slice[4096 + m0 + r], fsum[f]);
  }
}

// ---------------- Pass 2: out = (q' @ num) / max(q' @ den, EPS) -----------------
// wave task: (bh, 64-row q block). 4 iters x 16 q rows. LDS transpose of q'.
// All 8 waves share one bh; proj + state slice staged once per block via TDM.
__global__ void __launch_bounds__(256)
pass2(const float* __restrict__ qs, const float* __restrict__ proj,
      const float* __restrict__ bufc, float* __restrict__ out) {
  __shared__ __align__(16) _Float16 lds[8][16 * 72];
#if USE_TDM
  __shared__ __align__(16) float sproj[4096];
  __shared__ __align__(16) float sbuf[BUFSLICE];
#endif
  const int l = threadIdx.x & 31, wv = threadIdx.x >> 5;
  const int w = blockIdx.x * 8 + wv;
  const int sb = w & 127;          // 128 q blocks of 64 rows
  const int bh = w >> 7;           // uniform within the block
  const int b = bh >> 3, h = bh & 7;
  const int g = l >> 4, r = l & 15;
  _Float16* myl = lds[wv];

#if USE_TDM
  if (wv == 0) {
    tdm_copy_1d(proj, sproj, 4096);
    tdm_copy_1d(bufc + (size_t)bh * BUFSLICE, sbuf, BUFSLICE);
    __builtin_amdgcn_s_wait_tensorcnt(0);
  }
  __syncthreads();
  const float* psrc = sproj;
  const float* bsrc = sbuf;
#else
  const float* psrc = proj;
  const float* bsrc = bufc + (size_t)bh * BUFSLICE;
#endif

  v16h pj[2][4];
#pragma unroll
  for (int s = 0; s < 2; ++s)
#pragma unroll
    for (int f = 0; f < 4; ++f) pj[s][f] = load_projT_frag(psrc, s, 16 * f, l);

  // state B-fragments: lane l = column l%16, feature rows 32*s+8*g+i (+16 for halves 8..15).
  v16h pb[2][4], pd[2];
#pragma unroll
  for (int s = 0; s < 2; ++s) {
    const float* brow = bsrc + (32 * s) * 64 + r;
#pragma unroll
    for (int n = 0; n < 4; ++n)
#pragma unroll
      for (int i = 0; i < 16; ++i) pb[s][n][i] = (_Float16)brow[(i < 8 ? 8 * g + i : 8 + 8 * g + i) * 64 + 16 * n];
    v16h d;
#pragma unroll
    for (int i = 0; i < 16; ++i) d[i] = (_Float16)(r == 0 ? bsrc[4096 + 32 * s + (i < 8 ? 8 * g + i : 8 + 8 * g + i)] : 0.0f);
    pd[s] = d;
  }

  const float* qbase = qs + (((size_t)b * SS) * HHN + h) * DD;
  float* obase = out + (((size_t)b * SS) * HHN + h) * DD;

  for (int it = 0; it < 4; ++it) {
    const int s0 = sb * 64 + it * 16;
    if (it < 3)
      __builtin_prefetch(qbase + (size_t)(s0 + 16 + r) * ROWSTRIDE, 0, 1);

    const float* xrow = qbase + (size_t)(s0 + r) * ROWSTRIDE;
    float ssq = 0.0f;
    v16h a0 = load_x_frag(xrow, 0, g, ssq);
    v16h a1 = load_x_frag(xrow, 1, g, ssq);
    const float hfull = HSCL * (ssq + lane_bcast(ssq, l ^ 16));
    float hrow[8];
#pragma unroll
    for (int j = 0; j < 8; ++j) hrow[j] = lane_bcast(hfull, j + 8 * g);

#pragma unroll
    for (int f = 0; f < 4; ++f) {
      v8f c = {};
      c = wmma16(a0, pj[0][f], c);
      c = wmma16(a1, pj[1][f], c);
#pragma unroll
      for (int j = 0; j < 8; ++j) {
        float v = __expf(c[j] - hrow[j]) * RSQM;
        myl[(j + 8 * g) * 72 + 16 * f + r] = (_Float16)v;  // [qrow][feature]
      }
    }
    // q' A-fragments: row r, features 32*s+8*g+(0..7) and +16 (two b128 reads).
    v16h aq[2];
#pragma unroll
    for (int s = 0; s < 2; ++s) {
      const _Float16* rp = myl + r * 72 + 32 * s + 8 * g;
      v8h lo = *(const v8h*)rp;
      v8h hi = *(const v8h*)(rp + 16);
#pragma unroll
      for (int i = 0; i < 8; ++i) {
        aq[s][i] = lo[i];
        aq[s][8 + i] = hi[i];
      }
    }
    v8f dacc = {};
    dacc = wmma16(aq[0], pd[0], dacc);
    dacc = wmma16(aq[1], pd[1], dacc);
    float den[8];
#pragma unroll
    for (int j = 0; j < 8; ++j)
      den[j] = fmaxf(lane_bcast(dacc[j], l & 16), EPSV);
#pragma unroll
    for (int n = 0; n < 4; ++n) {
      v8f o = {};
      o = wmma16(aq[0], pb[0][n], o);
      o = wmma16(aq[1], pb[1][n], o);
#pragma unroll
      for (int j = 0; j < 8; ++j)
        obase[(size_t)(s0 + j + 8 * g) * ROWSTRIDE + 16 * n + r] = o[j] / den[j];
    }
  }
}

extern "C" void kernel_launch(void* const* d_in, const int* in_sizes, int n_in,
                              void* d_out, int out_size, void* d_ws, size_t ws_size,
                              hipStream_t stream) {
  (void)in_sizes; (void)n_in; (void)out_size; (void)ws_size;
  const float* qs = (const float*)d_in[0];
  const float* ks = (const float*)d_in[1];
  const float* vs = (const float*)d_in[2];
  const float* proj = (const float*)d_in[3];
  float* bufc = (float*)d_ws;                 // [32][4160] f32 (num 64x64 | den 64)
  const int nz = NBH * BUFSLICE;

  zerok<<<(nz + 255) / 256, 256, 0, stream>>>(bufc, nz);
  pass1<<<512, 256, 0, stream>>>(ks, vs, proj, bufc);
  pass2<<<512, 256, 0, stream>>>(qs, proj, bufc, (float*)d_out);
}
